// SpatialRefinement_16535624089642
// MI455X (gfx1250) — matrix operand order repaired and checked by reading, not compiled, not run on hardware
//
#include <hip/hip_runtime.h>

// Problem constants (match reference)
#define HDIM  512
#define WDIM  512
#define BATCH 32
#define NLANDM 100000
#define NF    16
#define HWSZ  (HDIM * WDIM)
#define EPSV  1e-3f

typedef __attribute__((ext_vector_type(16))) _Float16 v16h;
typedef __attribute__((ext_vector_type(8)))  _Float16 v8h;
typedef __attribute__((ext_vector_type(8)))  float    v8f;

// ---------------- K1: zero the scatter grid ----------------
__global__ void zero_kernel(float* __restrict__ p, long n) {
    long i = (long)blockIdx.x * blockDim.x + threadIdx.x;
    long stride = (long)gridDim.x * blockDim.x;
    for (; i < n; i += stride) p[i] = 0.0f;
}

// ---------------- K2: scatter-add landmarks into grid ----------------
__global__ void scatter_kernel(const float* __restrict__ x,
                               const int* __restrict__ fidx,
                               float* __restrict__ grid) {
    long i = (long)blockIdx.x * blockDim.x + threadIdx.x;
    long total = (long)BATCH * NLANDM;
    if (i >= total) return;
    int b = (int)(i / NLANDM);
    int j = (int)(i - (long)b * NLANDM);
    atomicAdd(&grid[(long)b * HWSZ + fidx[j]], x[i]);
}

// ---------------- K3: fused conv1 + ReLU + BN + conv2 (WMMA) ----------------
// One 32x32 output tile per workgroup (8 wave32).
//  Stage A: 36x36 grid tile (halo 2) -> f16 LDS
//  Stage B: conv1 over 34x34 region (halo 1) via v_wmma_f32_16x16x32_f16 (K=9 padded to 32)
//  Stage C: conv2 over 32x32 interior via 5 accumulating WMMAs (K=144 padded to 160);
//           A fragments assembled with 2x ds_load_b128 per WMMA.
__launch_bounds__(256)
__global__ void conv_fused_kernel(const float* __restrict__ grid,
                                  const float* __restrict__ w1,    // (3,3,1,16) -> [k*16+c]
                                  const float* __restrict__ b1,    // (16)
                                  const float* __restrict__ gammav,
                                  const float* __restrict__ betav,
                                  const float* __restrict__ mmean,
                                  const float* __restrict__ mvar,
                                  const float* __restrict__ w2,    // (3,3,16,1) -> [k*16+c]
                                  const float* __restrict__ b2s,   // (1)
                                  float* __restrict__ refined) {
    __shared__ _Float16 gt[36 * 36];          // grid tile, halo 2
    __shared__ _Float16 y1[34 * 34 * NF];     // conv1+BN result, halo 1

    const int tx   = blockIdx.x;              // W tile
    const int ty   = blockIdx.y;              // H tile
    const int b    = blockIdx.z;              // batch
    const int tid  = threadIdx.x;
    const int lane = tid & 31;
    const int wave = tid >> 5;
    const int half = (lane >> 4) & 1;
    const int ln   = lane & 15;

    const float* gbase = grid + (long)b * HWSZ;

    // ---- Stage A: cooperative load of 36x36 tile with SAME-padding ----
    for (int i = tid; i < 36 * 36; i += 256) {
        int gy = i / 36, gx = i - gy * 36;
        int Y = ty * 32 - 2 + gy;
        int X = tx * 32 - 2 + gx;
        float v = 0.0f;
        if (Y >= 0 && Y < HDIM && X >= 0 && X < WDIM) v = gbase[Y * WDIM + X];
        gt[i] = (_Float16)v;
    }

    // Per-lane channel constants (channel = ln for C/D layout N = lane&15)
    float scale = gammav[ln] * rsqrtf(mvar[ln] + EPSV);
    float shift = betav[ln] - mmean[ln] * scale;
    float bias  = b1[ln];

    // conv1 B fragment: element e holds (K = e + (e>=8 ? 8 : 0) + 8*half, N = ln); K>=9 padded 0
    v16h bf1;
#pragma unroll
    for (int e = 0; e < 16; ++e) {
        int K = e + ((e >= 8) ? 8 : 0) + half * 8;
        bf1[e] = (K < 9) ? (_Float16)w1[K * NF + ln] : (_Float16)0.0f;
    }

    __syncthreads();

    // ---- Stage B: conv1+ReLU+BN over 34x34 = 1156 pixels (73 tiles of 16) ----
    for (int t = wave; t < 73; t += 8) {
        int base = t * 16;
        int m = base + ln; if (m > 1155) m = 1155;      // clamp partial tile
        int py = m / 34, px = m - py * 34;
        int rowbase = py * 36 + px;                      // loop-invariant LDS base
        // A fragment: 16-bit A 16x32 layout: K = e + (e>=8 ? 8 : 0) + 8*half
        v16h af;
#pragma unroll
        for (int e = 0; e < 16; ++e) {
            int K = e + ((e >= 8) ? 8 : 0) + half * 8;
            _Float16 v = (_Float16)0.0f;
            if (K < 9) {
                int ky = K / 3, kx = K - ky * 3;
                v = gt[rowbase + ky * 36 + kx];          // const offsets -> ds imm
            }
            af[e] = v;
        }
        v8f c = {};
        c = __builtin_amdgcn_wmma_f32_16x16x32_f16(false, af, false, bf1,
                                                   (short)0, c, false, false);
        // C layout: VGPR r -> pixel M = r + 8*half, channel N = ln
#pragma unroll
        for (int r = 0; r < 8; ++r) {
            int m2 = base + r + half * 8; if (m2 > 1155) m2 = 1155;
            float z = c[r] + bias;
            z = (z > 0.0f) ? z : 0.0f;       // ReLU
            z = z * scale + shift;            // inference BN
            y1[m2 * NF + ln] = (_Float16)z;
        }
    }

    __syncthreads();

    // conv2 B fragments: logical Kg = k*16 + c = flat index into w2; only N==0 valid
    v16h bf2[5];
#pragma unroll
    for (int kk = 0; kk < 5; ++kk) {
#pragma unroll
        for (int e = 0; e < 16; ++e) {
            int Kg = kk * 32 + e + ((e >= 8) ? 8 : 0) + half * 8;
            bf2[kk][e] = (ln == 0 && Kg < 144) ? (_Float16)w2[Kg] : (_Float16)0.0f;
        }
    }
    float cb = b2s[0];

    // ---- Stage C: conv2 over 32x32 = 1024 pixels (64 tiles of 16) ----
    // For chunk kk: A elements e=0..7  = tap 2kk,   channels 8*half + (0..7)
    //              A elements e=8..15 = tap 2kk+1, channels 8*half + (0..7)
    // Each is 8 contiguous f16 in y1 (16B-aligned) -> one ds_load_b128.
    for (int t = wave; t < 64; t += 8) {
        int base = t * 16;
        int m = base + ln;
        int oy = m >> 5, ox = m & 31;
        int pixbase = (oy * 34 + ox) * NF + 8 * half;    // lane's LDS element base
        v8f c = {};
#pragma unroll
        for (int kk = 0; kk < 5; ++kk) {
            const int tap0 = 2 * kk;
            const int ky0 = tap0 / 3, kx0 = tap0 - ky0 * 3;
            v8h lo = *(const v8h*)&y1[pixbase + (ky0 * 34 + kx0) * NF];
            v8h hi = {};
            if (kk < 4) {                                // tap 2kk+1 valid (tap 9 padded)
                const int tap1 = 2 * kk + 1;
                const int ky1 = tap1 / 3, kx1 = tap1 - ky1 * 3;
                hi = *(const v8h*)&y1[pixbase + (ky1 * 34 + kx1) * NF];
            }
            v16h af;
#pragma unroll
            for (int e = 0; e < 8; ++e) { af[e] = lo[e]; af[e + 8] = hi[e]; }
            c = __builtin_amdgcn_wmma_f32_16x16x32_f16(false, af, false, bf2[kk],
                                                       (short)0, c, false, false);
        }
        // Only column N==0 holds the single output channel (lanes 0 and 16);
        // each writes 8 consecutive floats -> 2x global_store_b128.
        if (ln == 0) {
            int oy2 = base >> 5;
            int ox2 = (base & 31) + half * 8;
            float* dst = refined + (long)b * HWSZ +
                         (long)(ty * 32 + oy2) * WDIM + (tx * 32 + ox2);
#pragma unroll
            for (int r = 0; r < 8; ++r) dst[r] = c[r] + cb;
        }
    }
}

// ---------------- K4: gather refined values and add to x ----------------
__global__ void gather_kernel(const float* __restrict__ x,
                              const int* __restrict__ fidx,
                              const float* __restrict__ refined,
                              float* __restrict__ out) {
    long i = (long)blockIdx.x * blockDim.x + threadIdx.x;
    long total = (long)BATCH * NLANDM;
    if (i >= total) return;
    int b = (int)(i / NLANDM);
    int j = (int)(i - (long)b * NLANDM);
    out[i] = x[i] + refined[(long)b * HWSZ + fidx[j]];
}

extern "C" void kernel_launch(void* const* d_in, const int* in_sizes, int n_in,
                              void* d_out, int out_size, void* d_ws, size_t ws_size,
                              hipStream_t stream) {
    const float* x      = (const float*)d_in[0];
    const float* w1     = (const float*)d_in[1];
    const float* b1     = (const float*)d_in[2];
    const float* gammav = (const float*)d_in[3];
    const float* betav  = (const float*)d_in[4];
    const float* mmean  = (const float*)d_in[5];
    const float* mvar   = (const float*)d_in[6];
    const float* w2     = (const float*)d_in[7];
    const float* b2     = (const float*)d_in[8];
    const int*   fidx   = (const int*)d_in[9];
    float* out = (float*)d_out;

    float* grid    = (float*)d_ws;                    // B*HW floats = 33.5 MB
    float* refined = grid + (long)BATCH * HWSZ;       // B*HW floats = 33.5 MB

    long ngrid = (long)BATCH * HWSZ;
    zero_kernel<<<1024, 256, 0, stream>>>(grid, ngrid);

    long nsc = (long)BATCH * NLANDM;
    scatter_kernel<<<(int)((nsc + 255) / 256), 256, 0, stream>>>(x, fidx, grid);

    dim3 g(WDIM / 32, HDIM / 32, BATCH);
    conv_fused_kernel<<<g, 256, 0, stream>>>(grid, w1, b1, gammav, betav,
                                             mmean, mvar, w2, b2, refined);

    gather_kernel<<<(int)((nsc + 255) / 256), 256, 0, stream>>>(x, fidx, refined, out);
}
